// VectorQuantizer_1795296330062
// MI455X (gfx1250) — matrix operand order repaired and checked by reading, not compiled, not run on hardware
//
#include <hip/hip_runtime.h>
#include <hip/hip_bf16.h>

typedef __attribute__((ext_vector_type(16))) _Float16 v16h;
typedef __attribute__((ext_vector_type(8)))  _Float16 v8h;
typedef __attribute__((ext_vector_type(8)))  float    v8f;

#define NUM_CODES 512
#define DIM 64
#define N_PIX (32 * 64 * 64)          // 131072
#define TILE_PX 128
#define XT_STRIDE 68                  // floats; 272B rows: 16B-aligned, few bank conflicts
#define Q_ELEMS (32 * 64 * 64 * 64)   // 8388608

// ---------------- prep: f16 codebook, -0.5*||w||^2, zero loss accumulator ----
__global__ __launch_bounds__(256) void vq_prep(const float* __restrict__ w,
                                               _Float16* __restrict__ wf16,
                                               float* __restrict__ cneg,
                                               float* __restrict__ acc) {
    int k = blockIdx.x * 256 + threadIdx.x;   // grid = 2 blocks -> 512 codes
    if (k == 0) *acc = 0.0f;
    if (k < NUM_CODES) {
        const float* row = w + k * DIM;
        _Float16* orow = wf16 + k * DIM;
        float s = 0.0f;
#pragma unroll 8
        for (int c = 0; c < DIM; ++c) {
            float v = row[c];
            s += v * v;
            orow[c] = (_Float16)v;
        }
        cneg[k] = -0.5f * s;
    }
}

// ---------------- main: WMMA argmax + fused gather / store / loss -----------
__global__ __launch_bounds__(256) void vq_main(const float* __restrict__ in,
                                               const float* __restrict__ w,
                                               const _Float16* __restrict__ wf16,
                                               const float* __restrict__ cneg,
                                               float* __restrict__ out_q,
                                               float* __restrict__ out_idx,
                                               float* __restrict__ acc) {
    __shared__ alignas(16) float xt[TILE_PX * XT_STRIDE];   // f32 pixel tile, transposed
    __shared__ int   idxs[TILE_PX];
    __shared__ float red[8];

    const int tile = blockIdx.x;            // 1024 tiles of 128 pixels
    const int P0   = tile * TILE_PX;
    const int b    = P0 >> 12;              // / (64*64)
    const int hw0  = P0 & 4095;
    const float* inb  = in    + (size_t)b * DIM * 4096 + hw0;
    float*       outb = out_q + (size_t)b * DIM * 4096 + hw0;
    const int t = threadIdx.x;

    // Stage 64ch x 128px tile (coalesced, non-temporal: keep WGP$ for codebook),
    // transposed to pixel-major f32 in LDS. Read input exactly once.
#pragma unroll
    for (int i = 0; i < 32; ++i) {
        int e = i * 256 + t;
        int c = e >> 7, p = e & 127;
        float v = __builtin_nontemporal_load(&inb[(size_t)c * 4096 + p]);
        xt[p * XT_STRIDE + c] = v;
    }
    __syncthreads();

    const int wave = t >> 5, lane = t & 31;
    const int col  = lane & 15;             // pixel column / code row within tile
    const bool hi  = lane >= 16;
    const int px   = wave * 16 + col;       // pixel within 128-px tile
    const int mo   = hi ? 8  : 0;           // A K-shift & C/D M-shift for hi lanes

    // B fragments (X^T columns): 32 f32 from LDS -> f16, loaded once per wave.
    const float4* xp = reinterpret_cast<const float4*>(&xt[px * XT_STRIDE + (hi ? 8 : 0)]);
    float4 f0 = xp[0], f1 = xp[1], f2 = xp[4], f3 = xp[5];   // K 0..7, 16..23 (+kb)
    float4 g0 = xp[8], g1 = xp[9], g2 = xp[12], g3 = xp[13]; // K 32..39, 48..55 (+kb)
    v16h b0, b1;
#pragma unroll
    for (int i = 0; i < 4; ++i) {
        b0[i]      = (_Float16)(&f0.x)[i];  b0[i + 4]  = (_Float16)(&f1.x)[i];
        b0[i + 8]  = (_Float16)(&f2.x)[i];  b0[i + 12] = (_Float16)(&f3.x)[i];
        b1[i]      = (_Float16)(&g0.x)[i];  b1[i + 4]  = (_Float16)(&g1.x)[i];
        b1[i + 8]  = (_Float16)(&g2.x)[i];  b1[i + 12] = (_Float16)(&g3.x)[i];
    }

    float best = -3.4e38f;
    int bestIdx = 0;

    for (int ct = 0; ct < 32; ++ct) {
        const _Float16* wr = wf16 + (ct * 16 + col) * DIM;   // A row: code ct*16+col
        v16h a0, a1;
        {
            v8h p0 = *reinterpret_cast<const v8h*>(wr + mo);
            v8h p1 = *reinterpret_cast<const v8h*>(wr + 16 + mo);
            v8h p2 = *reinterpret_cast<const v8h*>(wr + 32 + mo);
            v8h p3 = *reinterpret_cast<const v8h*>(wr + 48 + mo);
#pragma unroll
            for (int i = 0; i < 8; ++i) {
                a0[i] = p0[i]; a0[i + 8] = p1[i];
                a1[i] = p2[i]; a1[i + 8] = p3[i];
            }
        }
        v8f cacc;
        const float* cp = cneg + ct * 16 + mo;               // bias -||w||^2/2
#pragma unroll
        for (int r = 0; r < 8; ++r) cacc[r] = cp[r];

        cacc = __builtin_amdgcn_wmma_f32_16x16x32_f16(false, a0, false, b0,
                                                      (short)0, cacc, false, false);
        cacc = __builtin_amdgcn_wmma_f32_16x16x32_f16(false, a1, false, b1,
                                                      (short)0, cacc, false, false);
#pragma unroll
        for (int r = 0; r < 8; ++r) {
            float v = cacc[r];
            if (v > best) { best = v; bestIdx = ct * 16 + mo + r; }
        }
    }

    // combine lane halves (codes m and m+8 of each tile live 16 lanes apart)
    float ob = __shfl_xor(best, 16, 32);
    int   oi = __shfl_xor(bestIdx, 16, 32);
    if (ob > best || (ob == best && oi < bestIdx)) { best = ob; bestIdx = oi; }
    if (!hi) {
        idxs[px] = bestIdx;
        out_idx[P0 + px] = (float)bestIdx;
    }
    __syncthreads();

    // fused gather (codebook stays hot in WGP$/L2) + NCHW store + loss partial;
    // x re-read from LDS f32 (no second HBM pass).
    float ls = 0.0f;
#pragma unroll
    for (int i = 0; i < 32; ++i) {
        int e = i * 256 + t;
        int c = e >> 7, p = e & 127;
        float x = xt[p * XT_STRIDE + c];
        float q = w[idxs[p] * DIM + c];
        __builtin_nontemporal_store(q, &outb[(size_t)c * 4096 + p]);  // quantized_st == q
        float d = q - x;
        ls += d * d;
    }
    // wave-level reduction, then tiny LDS combine
#pragma unroll
    for (int s = 16; s > 0; s >>= 1) ls += __shfl_down(ls, s, 32);
    if (lane == 0) red[wave] = ls;
    __syncthreads();
    if (t == 0) {
        float s0 = 0.0f;
#pragma unroll
        for (int i = 0; i < 8; ++i) s0 += red[i];
        atomicAdd(acc, s0);
    }
}

// ---------------- finalize scalar loss --------------------------------------
__global__ void vq_loss(const float* __restrict__ acc, float* __restrict__ out_loss) {
    if (threadIdx.x == 0)
        out_loss[0] = 1.25f * acc[0] / (float)Q_ELEMS;  // q_loss + 0.25*e_loss
}

extern "C" void kernel_launch(void* const* d_in, const int* in_sizes, int n_in,
                              void* d_out, int out_size, void* d_ws, size_t ws_size,
                              hipStream_t stream) {
    const float* inputs = (const float*)d_in[0];   // [32,64,64,64] f32
    const float* weight = (const float*)d_in[1];   // [512,64] f32
    float* out = (float*)d_out;                    // [Q | loss | idx]

    char* ws = (char*)d_ws;
    float*    acc  = (float*)ws;                   // 4B (zeroed in vq_prep)
    _Float16* wf16 = (_Float16*)(ws + 256);        // 512*64*2 = 64 KB
    float*    cneg = (float*)(ws + 256 + 65536);   // 512*4 = 2 KB

    vq_prep<<<2, 256, 0, stream>>>(weight, wf16, cneg, acc);
    vq_main<<<N_PIX / TILE_PX, 256, 0, stream>>>(inputs, weight, wf16, cneg,
                                                 out, out + (Q_ELEMS + 1), acc);
    vq_loss<<<1, 32, 0, stream>>>(acc, out + Q_ELEMS);
}
